// ANR_79980880986127
// MI455X (gfx1250) — matrix operand order repaired and checked by reading, not compiled, not run on hardware
//
#include <hip/hip_runtime.h>
#include <hip/hip_bf16.h>

// ---------------- problem constants (match reference) ----------------
#define A_N   5      // aspects
#define L_N   500    // doc length
#define D_N   300    // word dim
#define H1_N  10
#define H2_N  50
#define B_N   256    // batch
#define NT    4      // N tiles (50 cols padded to 64)
#define KPAD  328    // LDS row stride in halves (656B, 16B-aligned, bank-skewed)

typedef __attribute__((ext_vector_type(16))) _Float16 v16h;
typedef __attribute__((ext_vector_type(8)))  _Float16 v8h;
typedef __attribute__((ext_vector_type(8)))  float    v8f;
typedef __attribute__((ext_vector_type(4)))  float    v4f;

// =====================================================================
// Kernel 1: fused gather + projection GEMM via WMMA
//   per doc g in [0, 2B): C[500,50] = E[500,300] x P[300,50]
//   E rows gathered from words_emb (60 MB -> L2-resident on MI455X)
// =====================================================================
__global__ void __launch_bounds__(256)
anr_proj_gemm(const int* __restrict__ U_ids, const int* __restrict__ I_ids,
              const int* __restrict__ U_docs, const int* __restrict__ I_docs,
              const float* __restrict__ wemb, const float* __restrict__ aproj,
              float* __restrict__ adoc_ws)
{
    __shared__ _Float16 Bsh[64 * KPAD];   // P^T, f16, zero-padded: [n][k]

    const int tid  = threadIdx.x;
    const int g    = blockIdx.x;          // 0..2B-1
    const int side = g >> 8;              // 0 = user, 1 = item
    const int b    = g & (B_N - 1);

    // stage projection matrix into LDS: Bsh[n][k] = aspect_proj[n/10][k][n%10]
    for (int idx = tid; idx < 64 * 320; idx += 256) {
        const int n = idx / 320, k = idx % 320;
        float v = 0.f;
        if (n < A_N * H1_N && k < D_N)
            v = aproj[(n / H1_N) * (D_N * H1_N) + k * H1_N + (n % H1_N)];
        Bsh[n * KPAD + k] = (_Float16)v;
    }
    __syncthreads();

    const int* __restrict__ doc =
        (side == 0) ? (U_docs + (size_t)U_ids[b] * L_N)
                    : (I_docs + (size_t)I_ids[b] * L_N);

    const int lane   = tid & 31;
    const int wave   = tid >> 5;
    const int laneLo = lane & 15;
    const int laneHi = lane >> 4;

    for (int mt = wave; mt < 32; mt += 8) {           // 32 M-tiles over 8 waves
        // clamp row index: padded rows (m>=500) compute garbage that is
        // never stored, so no zeroing/predication needed on A.
        const int m   = mt * 16 + laneLo;
        const int mc  = (m < L_N) ? m : (L_N - 1);
        const int wid = doc[mc];
        const float* __restrict__ row = wemb + (size_t)wid * D_N;

        v8f acc[NT];
        #pragma unroll
        for (int nt = 0; nt < NT; ++nt) acc[nt] = (v8f){};

        // ---- K-steps 0..8: all loads in-bounds (max K = 287 < 300) ----
        for (int ks = 0; ks < 9; ++ks) {
            __builtin_prefetch(row + (ks + 1) * 32, 0, 1);

            // A 16x32 f16 layout: halves 0..7 = K0..K0+7, 8..15 = K0+16..K0+23
            const int K0 = ks * 32 + laneHi * 8;
            const v4f r0a = *(const v4f*)(row + K0);
            const v4f r0b = *(const v4f*)(row + K0 + 4);
            const v4f r1a = *(const v4f*)(row + K0 + 16);
            const v4f r1b = *(const v4f*)(row + K0 + 20);

            v16h afrag;
            #pragma unroll
            for (int i = 0; i < 4; ++i) {
                afrag[i]      = (_Float16)r0a[i];
                afrag[4 + i]  = (_Float16)r0b[i];
                afrag[8 + i]  = (_Float16)r1a[i];
                afrag[12 + i] = (_Float16)r1b[i];
            }

            const int koff = ks * 32 + (laneHi << 3);
            #pragma unroll
            for (int nt = 0; nt < NT; ++nt) {
                const _Float16* bp = &Bsh[(nt * 16 + laneLo) * KPAD + koff];
                const v8h lo = *(const v8h*)bp;
                const v8h hi = *(const v8h*)(bp + 16);
                v16h bfrag;
                #pragma unroll
                for (int i = 0; i < 8; ++i) { bfrag[i] = lo[i]; bfrag[8 + i] = hi[i]; }
                acc[nt] = __builtin_amdgcn_wmma_f32_16x16x32_f16(
                    false, afrag, false, bfrag, (short)0, acc[nt], false, false);
            }
        }

        // ---- K-step 9 (K = 288..319, valid only K < 300), branch-free ----
        {
            // laneHi=0 needs 288..295 (valid) ; laneHi=1 needs 296..303 -> 296..299
            const v4f r0a = *(const v4f*)(row + 288 + laneHi * 8); // in-bounds
            const v4f t   = *(const v4f*)(row + 292);              // in-bounds
            const v4f zero = (v4f){};
            const v4f r0b = laneHi ? zero : t;                     // 300.. -> 0

            v16h afrag;
            #pragma unroll
            for (int i = 0; i < 4; ++i) {
                afrag[i]      = (_Float16)r0a[i];
                afrag[4 + i]  = (_Float16)r0b[i];
                afrag[8 + i]  = (_Float16)0.f;   // K = 304.. / 312.. all >= 300
                afrag[12 + i] = (_Float16)0.f;
            }

            const int koff = 9 * 32 + (laneHi << 3);
            #pragma unroll
            for (int nt = 0; nt < NT; ++nt) {
                const _Float16* bp = &Bsh[(nt * 16 + laneLo) * KPAD + koff];
                const v8h lo = *(const v8h*)bp;
                const v8h hi = *(const v8h*)(bp + 16);
                v16h bfrag;
                #pragma unroll
                for (int i = 0; i < 8; ++i) { bfrag[i] = lo[i]; bfrag[8 + i] = hi[i]; }
                acc[nt] = __builtin_amdgcn_wmma_f32_16x16x32_f16(
                    false, afrag, false, bfrag, (short)0, acc[nt], false, false);
            }
        }

        // scatter C: VGPR r -> M = r + 8*(lane>=16); lane&15 -> N
        #pragma unroll
        for (int nt = 0; nt < NT; ++nt) {
            #pragma unroll
            for (int r = 0; r < 8; ++r) {
                const int mm = mt * 16 + r + laneHi * 8;
                const int n  = nt * 16 + laneLo;
                if (mm < L_N && n < A_N * H1_N) {
                    const int a = n / H1_N, hh = n % H1_N;
                    adoc_ws[(((size_t)g * A_N + a) * L_N + mm) * H1_N + hh] = acc[nt][r];
                }
            }
        }
    }
}

// =====================================================================
// Kernel 2: context-window attention + softmax over L + weighted reduce
//   one block per (doc g, aspect a)
// =====================================================================
__global__ void __launch_bounds__(256)
anr_attn(const float* __restrict__ adoc_ws, const float* __restrict__ aemb,
         float* __restrict__ rep_ws)
{
    __shared__ float sdoc[L_N * H1_N];     // 20 KB
    __shared__ float attnS[512];
    __shared__ float red[256];
    __shared__ float preps[256 * H1_N];    // 10 KB

    const int tid = threadIdx.x;
    const int a   = blockIdx.x % A_N;
    const int g   = blockIdx.x / A_N;
    const float* __restrict__ src = adoc_ws + ((size_t)g * A_N + a) * L_N * H1_N;

    for (int i = tid; i < L_N * H1_N; i += 256) sdoc[i] = src[i];
    __syncthreads();

    float ae[30];
    #pragma unroll
    for (int i = 0; i < 30; ++i) ae[i] = aemb[a * 30 + i];

    float lg[2];
    float lmax = -1e30f;
    #pragma unroll
    for (int t = 0; t < 2; ++t) {
        const int l = tid + t * 256;
        float s = -1e30f;
        if (l < L_N) {
            s = 0.f;
            #pragma unroll
            for (int c = 0; c < 3; ++c) {
                const int lw = l + c - 1;              // pad = 1
                if (lw >= 0 && lw < L_N) {
                    #pragma unroll
                    for (int h = 0; h < H1_N; ++h)
                        s += sdoc[lw * H1_N + h] * ae[c * H1_N + h];
                }
            }
            lmax = fmaxf(lmax, s);
        }
        lg[t] = s;
    }

    // block max
    red[tid] = lmax; __syncthreads();
    for (int s = 128; s > 0; s >>= 1) {
        if (tid < s) red[tid] = fmaxf(red[tid], red[tid + s]);
        __syncthreads();
    }
    const float mx = red[0];
    __syncthreads();

    float lsum = 0.f;
    #pragma unroll
    for (int t = 0; t < 2; ++t) {
        const int l = tid + t * 256;
        float e = 0.f;
        if (l < L_N) e = __expf(lg[t] - mx);
        attnS[l] = e;                                   // l < 512 always
        lsum += e;
    }
    red[tid] = lsum; __syncthreads();
    for (int s = 128; s > 0; s >>= 1) {
        if (tid < s) red[tid] += red[tid + s];
        __syncthreads();
    }
    const float inv = 1.f / red[0];

    // rep[h] = sum_l attn[l] * a_doc[l][h]
    float pr[H1_N];
    #pragma unroll
    for (int h = 0; h < H1_N; ++h) pr[h] = 0.f;
    #pragma unroll
    for (int t = 0; t < 2; ++t) {
        const int l = tid + t * 256;
        if (l < L_N) {
            const float w = attnS[l] * inv;
            #pragma unroll
            for (int h = 0; h < H1_N; ++h) pr[h] += w * sdoc[l * H1_N + h];
        }
    }
    #pragma unroll
    for (int h = 0; h < H1_N; ++h) preps[tid * H1_N + h] = pr[h];
    __syncthreads();
    if (tid < H1_N) {
        float s = 0.f;
        for (int t = 0; t < 256; ++t) s += preps[t * H1_N + tid];
        rep_ws[((size_t)g * A_N + a) * H1_N + tid] = s;
    }
}

// =====================================================================
// Kernel 3: co-attention + importance softmax + rating (one block per b)
// =====================================================================
__global__ void __launch_bounds__(64)
anr_tail(const float* __restrict__ rep_ws,
         const float* __restrict__ Mm, const float* __restrict__ up,
         const float* __restrict__ uw, const float* __restrict__ ip,
         const float* __restrict__ iw, const float* __restrict__ Bu,
         const float* __restrict__ Bi, const float* __restrict__ Bg,
         const int* __restrict__ U_ids, const int* __restrict__ I_ids,
         float* __restrict__ out)
{
    __shared__ float Ud[A_N * H1_N], Id[A_N * H1_N];
    __shared__ float aff[A_N * A_N];
    __shared__ float su[A_N], si[A_N], ar[A_N];

    const int tid = threadIdx.x;
    const int b   = blockIdx.x;
    const float* __restrict__ urp = rep_ws + (size_t)b * A_N * H1_N;
    const float* __restrict__ irp = rep_ws + ((size_t)B_N + b) * A_N * H1_N;

    if (tid < A_N * H1_N) { Ud[tid] = urp[tid]; Id[tid] = irp[tid]; }
    if (tid < A_N)        { su[tid] = 0.f; si[tid] = 0.f; }
    __syncthreads();

    // aff[a][c] = relu(Ud[a] . M . Id[c])
    if (tid < A_N * A_N) {
        const int a = tid / A_N, c = tid % A_N;
        float s = 0.f;
        for (int h = 0; h < H1_N; ++h) {
            float t = 0.f;
            for (int k = 0; k < H1_N; ++k) t += Mm[h * H1_N + k] * Id[c * H1_N + k];
            s += Ud[a * H1_N + h] * t;
        }
        aff[tid] = fmaxf(s, 0.f);
    }
    if (tid < A_N) {
        float s = 0.f;
        for (int h = 0; h < H1_N; ++h) s += Ud[tid * H1_N + h] * Id[tid * H1_N + h];
        ar[tid] = s;
    }
    __syncthreads();

    // thread e < H2: row of Hu1/Hi1, relu-mix, weighted accumulate (LDS atomics)
    if (tid < H2_N) {
        float hu1[A_N], hi1[A_N];
        for (int a = 0; a < A_N; ++a) {
            float s = 0.f, s2 = 0.f;
            for (int h = 0; h < H1_N; ++h) {
                s  += up[tid * H1_N + h] * Ud[a * H1_N + h];
                s2 += ip[tid * H1_N + h] * Id[a * H1_N + h];
            }
            hu1[a] = s; hi1[a] = s2;
        }
        for (int a = 0; a < A_N; ++a) {
            float tu = hu1[a];
            float ti = hi1[a];
            for (int c = 0; c < A_N; ++c)   tu += hi1[c]  * aff[a * A_N + c];   // Hi1 . aff^T
            for (int aa = 0; aa < A_N; ++aa) ti += hu1[aa] * aff[aa * A_N + a]; // Hu1 . aff
            atomicAdd(&su[a], uw[tid] * fmaxf(tu, 0.f));
            atomicAdd(&si[a], iw[tid] * fmaxf(ti, 0.f));
        }
    }
    __syncthreads();

    if (tid == 0) {
        float mu = su[0], mi = si[0];
        for (int a = 1; a < A_N; ++a) { mu = fmaxf(mu, su[a]); mi = fmaxf(mi, si[a]); }
        float eu[A_N], ei[A_N], zu = 0.f, zi = 0.f;
        for (int a = 0; a < A_N; ++a) {
            eu[a] = __expf(su[a] - mu); zu += eu[a];
            ei[a] = __expf(si[a] - mi); zi += ei[a];
        }
        float R = 0.f;
        for (int a = 0; a < A_N; ++a) R += (eu[a] / zu) * (ei[a] / zi) * ar[a];
        R += Bu[U_ids[b]] + Bi[I_ids[b]] + Bg[0];
        out[b] = R;
    }
}

// =====================================================================
extern "C" void kernel_launch(void* const* d_in, const int* in_sizes, int n_in,
                              void* d_out, int out_size, void* d_ws, size_t ws_size,
                              hipStream_t stream)
{
    (void)in_sizes; (void)n_in; (void)out_size; (void)ws_size;

    const int*   U_ids  = (const int*)  d_in[0];
    const int*   I_ids  = (const int*)  d_in[1];
    const int*   U_docs = (const int*)  d_in[2];
    const int*   I_docs = (const int*)  d_in[3];
    const float* wemb   = (const float*)d_in[4];
    const float* aemb   = (const float*)d_in[5];
    const float* aproj  = (const float*)d_in[6];
    const float* Mm     = (const float*)d_in[7];
    const float* up     = (const float*)d_in[8];
    const float* uw     = (const float*)d_in[9];
    const float* ip     = (const float*)d_in[10];
    const float* iw     = (const float*)d_in[11];
    const float* Bu     = (const float*)d_in[12];
    const float* Bi     = (const float*)d_in[13];
    const float* Bg     = (const float*)d_in[14];
    float*       out    = (float*)d_out;

    float* adoc = (float*)d_ws;                               // [2B][A][L][H1] f32
    float* rep  = adoc + (size_t)2 * B_N * A_N * L_N * H1_N;  // [2B][A][H1]   f32

    anr_proj_gemm<<<2 * B_N, 256, 0, stream>>>(U_ids, I_ids, U_docs, I_docs,
                                               wemb, aproj, adoc);
    anr_attn<<<2 * B_N * A_N, 256, 0, stream>>>(adoc, aemb, rep);
    anr_tail<<<B_N, 64, 0, stream>>>(rep, Mm, up, uw, ip, iw, Bu, Bi, Bg,
                                     U_ids, I_ids, out);
}
